// ContrastLossLocal_69930657514022
// MI455X (gfx1250) — matrix operand order repaired and checked by reading, not compiled, not run on hardware
//
#include <hip/hip_runtime.h>
#include <hip/hip_bf16.h>

typedef _Float16 v16h __attribute__((ext_vector_type(16)));
typedef _Float16 v8h  __attribute__((ext_vector_type(8)));
typedef float    v8f  __attribute__((ext_vector_type(8)));
typedef float    v16f __attribute__((ext_vector_type(16)));

#define A_N    4096
#define D_K    256
#define C_N    19
#define U_N    10
#define KM_N   10
#define NL_N   10000
#define P_N    1900
#define P_PAD  1904
#define PT_N   119          // P_PAD / 16
#define ZCH    25           // NL chunks
#define TPC    25           // 16-row N-tiles per chunk (25*25*16 = 10000)
#define INV_T  14.285714285714286f
#define NEG_BIG (-1.0e30f)

// ---- WMMA fragment builders (contiguous k = 16*half + s in both A and B, 16-bit A 16x32 / B 32x16) ----
__device__ __forceinline__ v16h frag_a(const _Float16* row, int kb, int half) {
    // A: lane m holds 16 contiguous K at kb + half*16
    v8h lo = *(const v8h*)(row + kb + half * 16);
    v8h hi = *(const v8h*)(row + kb + half * 16 + 8);
    return __builtin_shufflevector(lo, hi, 0,1,2,3,4,5,6,7,8,9,10,11,12,13,14,15);
}
__device__ __forceinline__ v16h frag_b(const _Float16* col, int kb, int half) {
    // B: lane n holds 16 contiguous K at kb + half*16
    v8h lo = *(const v8h*)(col + kb + half * 16);
    v8h hi = *(const v8h*)(col + kb + half * 16 + 8);
    return __builtin_shufflevector(lo, hi, 0,1,2,3,4,5,6,7,8,9,10,11,12,13,14,15);
}

// ---- CDNA5 async HBM->LDS (no VGPR landing, tracked by ASYNCcnt) ----
__device__ __forceinline__ void async_ld_b128(unsigned int lds_off, unsigned long long gaddr) {
    asm volatile("global_load_async_to_lds_b128 %0, %1, off"
                 :: "v"(lds_off), "v"(gaddr) : "memory");
}
__device__ __forceinline__ void wait_async_le32() {
    asm volatile("s_wait_asynccnt 0x20" ::: "memory");
}
__device__ __forceinline__ void wait_async_0() {
    asm volatile("s_wait_asynccnt 0x0" ::: "memory");
}

// ---------------- prep kernels ----------------
__global__ void k_zero(float* Nacc, int* counts) {
    int i = blockIdx.x * blockDim.x + threadIdx.x;
    if (i < A_N) Nacc[i] = 0.f;
    if (i < C_N) counts[i] = 0;
}

__global__ __launch_bounds__(256) void k_cvt_anchors(const float* __restrict__ a,
                                                     _Float16* __restrict__ ah) {
    size_t i = (size_t)blockIdx.x * 256 + threadIdx.x;
    ah[i] = (_Float16)a[i];
}

__global__ __launch_bounds__(256) void k_cvt_protos(const float* __restrict__ pm,
                                                    const int* __restrict__ pmask,
                                                    _Float16* __restrict__ ph,
                                                    int* __restrict__ pclass,
                                                    int* __restrict__ pvalid) {
    int p = blockIdx.x;
    int d = threadIdx.x;
    float v = (p < P_N) ? pm[(size_t)p * D_K + d] : 0.f;
    ph[(size_t)p * D_K + d] = (_Float16)v;
    if (d == 0) {
        if (p < P_N) {
            int cls = p / (U_N * KM_N);
            int u   = (p / KM_N) % U_N;
            pclass[p] = cls;
            pvalid[p] = pmask[cls * U_N + u];
        } else {
            pclass[p] = 0;
            pvalid[p] = 0;
        }
    }
}

__global__ void k_bucket(const int* __restrict__ labels, int* counts, int* lists) {
    int a = blockIdx.x * blockDim.x + threadIdx.x;
    if (a < A_N) {
        int c = labels[a];
        int pos = atomicAdd(&counts[c], 1);
        lists[c * A_N + pos] = a;
    }
}

__global__ void k_poscnt(const int* __restrict__ pmask, float* __restrict__ poscnt) {
    int c = threadIdx.x;
    if (c < C_N) {
        int s = 0;
        for (int u = 0; u < U_N; ++u) s += pmask[c * U_N + u];
        poscnt[c] = (float)(s * KM_N);
    }
}

// ---------------- anchors x protos : row max ----------------
__global__ __launch_bounds__(32) void k_protomax(const _Float16* __restrict__ ah,
                                                 const _Float16* __restrict__ ph,
                                                 const int* __restrict__ pvalid,
                                                 float* __restrict__ m) {
    __shared__ __align__(16) _Float16 Ap[16][D_K];
    __shared__ __align__(16) _Float16 Bp[16][D_K];
    const int lane  = threadIdx.x;
    const int atile = blockIdx.x;
    const int lm    = lane & 15;
    const int half  = lane >> 4;

    for (int i = 0; i < 16; ++i)
        ((uint4*)&Ap[i][0])[lane] = ((const uint4*)(ah + (size_t)(atile * 16 + i) * D_K))[lane];

    float rowmax[8];
#pragma unroll
    for (int r = 0; r < 8; ++r) rowmax[r] = NEG_BIG;

    for (int pt = 0; pt < PT_N; ++pt) {
        for (int i = 0; i < 16; ++i)
            ((uint4*)&Bp[i][0])[lane] = ((const uint4*)(ph + (size_t)(pt * 16 + i) * D_K))[lane];
        v8f c = {};
#pragma unroll
        for (int ks = 0; ks < 8; ++ks) {
            v16h av = frag_a(&Ap[lm][0], ks * 32, half);
            v16h bv = frag_b(&Bp[lm][0], ks * 32, half);
            c = __builtin_amdgcn_wmma_f32_16x16x32_f16(false, av, false, bv,
                                                       (short)0, c, false, false);
        }
        int pv = pvalid[pt * 16 + lm];
#pragma unroll
        for (int r = 0; r < 8; ++r) {
            float val = pv ? (c[r] * INV_T) : NEG_BIG;
            rowmax[r] = fmaxf(rowmax[r], val);
        }
    }
#pragma unroll
    for (int r = 0; r < 8; ++r) {
        float v = rowmax[r];
        v = fmaxf(v, __shfl_xor(v, 1));
        v = fmaxf(v, __shfl_xor(v, 2));
        v = fmaxf(v, __shfl_xor(v, 4));
        v = fmaxf(v, __shfl_xor(v, 8));
        if (lm == 0) m[atile * 16 + r + half * 8] = v;
    }
}

// ---------------- anchors x local memory (class-bucketed) : negatives sum ----------------
// One wave per block; B tiles streamed HBM->LDS with async DMA, double-buffered.
__global__ __launch_bounds__(32) void k_local(const float* __restrict__ local_mem,
                                              const _Float16* __restrict__ ah,
                                              const int* __restrict__ counts,
                                              const int* __restrict__ lists,
                                              const float* __restrict__ m,
                                              float* __restrict__ Nacc) {
    const int cls  = blockIdx.x;
    const int tile = blockIdx.y;
    const int n_anch = counts[cls];
    if (tile * 16 >= n_anch) return;
    const int zc = blockIdx.z;

    __shared__ __align__(16) _Float16 Ap[16][D_K];   // anchor panel (f16)        8 KB
    __shared__ __align__(16) float    Bf[2][16][D_K];// local panels (f32) 2x16KB 32 KB
    __shared__ float sm[16];
    __shared__ int   sidx[16];

    const int lane = threadIdx.x;
    const int lm   = lane & 15;
    const int half = lane >> 4;

    if (lane < 16) {
        int r = tile * 16 + lane;
        int a = (r < n_anch) ? lists[cls * A_N + r] : -1;
        sidx[lane] = a;
        sm[lane]   = (a >= 0) ? m[a] : 0.f;
    }
    // single wave: DS ops are in-order; compiler inserts dscnt waits.

    // stage A panel: 16 rows x 256 f16; one uint4 per lane per row
    for (int i = 0; i < 16; ++i) {
        int a = sidx[i];
        const uint4* src = (const uint4*)(ah + (size_t)((a >= 0) ? a : 0) * D_K);
        uint4 v = src[lane];
        if (a < 0) { v.x = v.y = v.z = v.w = 0u; }
        ((uint4*)&Ap[i][0])[lane] = v;
    }

    float mrow[8], accum[8];
#pragma unroll
    for (int r = 0; r < 8; ++r) { mrow[r] = sm[r + half * 8]; accum[r] = 0.f; }

    const unsigned long long gbase =
        (unsigned long long)(const char*)(local_mem + (size_t)cls * NL_N * D_K);
    const unsigned int bbase = (unsigned int)(uintptr_t)&Bf[0][0][0];
    const unsigned int lane16 = (unsigned int)(lane * 16);

    const int t0 = zc * TPC;
    const int t_end = t0 + TPC;

    // prologue: async-stage first tile (16 rows = contiguous 16 KB) into buffer 0
    {
        unsigned long long g = gbase + (unsigned long long)t0 * 16384ull;
#pragma unroll
        for (int j = 0; j < 32; ++j)
            async_ld_b128(bbase + j * 512 + lane16, g + j * 512 + lane16);
    }

    int buf = 0;
    for (int t = t0; t < t_end; ++t) {
        const bool pf = (t + 1 < t_end);
        if (pf) {   // prefetch next tile into the other buffer
            unsigned long long g = gbase + (unsigned long long)(t + 1) * 16384ull;
            unsigned int lb = bbase + (unsigned int)(buf ^ 1) * 16384u;
#pragma unroll
            for (int j = 0; j < 32; ++j)
                async_ld_b128(lb + j * 512 + lane16, g + j * 512 + lane16);
            wait_async_le32();   // in-order: current tile's 32 transfers complete
        } else {
            wait_async_0();
        }

        v8f c = {};
#pragma unroll
        for (int ks = 0; ks < 8; ++ks) {
            v16h av = frag_a(&Ap[lm][0], ks * 32, half);
            v16f bf = *(const v16f*)&Bf[buf][lm][ks * 32 + half * 16];
            v16h bv = __builtin_convertvector(bf, v16h);
            c = __builtin_amdgcn_wmma_f32_16x16x32_f16(false, av, false, bv,
                                                       (short)0, c, false, false);
        }
#pragma unroll
        for (int r = 0; r < 8; ++r)
            accum[r] += __expf(c[r] * INV_T - mrow[r]);
        buf ^= 1;
    }

#pragma unroll
    for (int r = 0; r < 8; ++r) {
        float v = accum[r];
        v += __shfl_xor(v, 1);
        v += __shfl_xor(v, 2);
        v += __shfl_xor(v, 4);
        v += __shfl_xor(v, 8);
        if (lm == 0) {
            int a = sidx[r + half * 8];
            if (a >= 0) atomicAdd(&Nacc[a], v);
        }
    }
}

// ---------------- anchors x protos : positive log-prob sums ----------------
__global__ __launch_bounds__(32) void k_protopos(const _Float16* __restrict__ ah,
                                                 const _Float16* __restrict__ ph,
                                                 const int* __restrict__ pvalid,
                                                 const int* __restrict__ pclass,
                                                 const int* __restrict__ labels,
                                                 const float* __restrict__ m,
                                                 const float* __restrict__ Nacc,
                                                 float* __restrict__ possum) {
    __shared__ __align__(16) _Float16 Ap[16][D_K];
    __shared__ __align__(16) _Float16 Bp[16][D_K];
    const int lane  = threadIdx.x;
    const int atile = blockIdx.x;
    const int lm    = lane & 15;
    const int half  = lane >> 4;

    for (int i = 0; i < 16; ++i)
        ((uint4*)&Ap[i][0])[lane] = ((const uint4*)(ah + (size_t)(atile * 16 + i) * D_K))[lane];

    float mr[8], Nr[8], rs[8];
    int lr[8];
#pragma unroll
    for (int r = 0; r < 8; ++r) {
        int a = atile * 16 + r + half * 8;
        mr[r] = m[a]; Nr[r] = Nacc[a]; lr[r] = labels[a]; rs[r] = 0.f;
    }

    for (int pt = 0; pt < PT_N; ++pt) {
        for (int i = 0; i < 16; ++i)
            ((uint4*)&Bp[i][0])[lane] = ((const uint4*)(ph + (size_t)(pt * 16 + i) * D_K))[lane];
        v8f c = {};
#pragma unroll
        for (int ks = 0; ks < 8; ++ks) {
            v16h av = frag_a(&Ap[lm][0], ks * 32, half);
            v16h bv = frag_b(&Bp[lm][0], ks * 32, half);
            c = __builtin_amdgcn_wmma_f32_16x16x32_f16(false, av, false, bv,
                                                       (short)0, c, false, false);
        }
        int col = pt * 16 + lm;
        int pv = pvalid[col];
        int pc = pclass[col];
#pragma unroll
        for (int r = 0; r < 8; ++r) {
            if (pv && pc == lr[r]) {
                float t = c[r] * INV_T - mr[r];
                rs[r] += t - __logf(__expf(t) + Nr[r]);
            }
        }
    }
#pragma unroll
    for (int r = 0; r < 8; ++r) {
        float v = rs[r];
        v += __shfl_xor(v, 1);
        v += __shfl_xor(v, 2);
        v += __shfl_xor(v, 4);
        v += __shfl_xor(v, 8);
        if (lm == 0) possum[atile * 16 + r + half * 8] = v;
    }
}

// ---------------- final reduction ----------------
__global__ __launch_bounds__(256) void k_final(const float* __restrict__ possum,
                                               const int* __restrict__ labels,
                                               const float* __restrict__ poscnt,
                                               float* __restrict__ out) {
    __shared__ float red[256];
    float s = 0.f;
    for (int a = threadIdx.x; a < A_N; a += 256)
        s += possum[a] / poscnt[labels[a]];
    red[threadIdx.x] = s;
    __syncthreads();
    for (int off = 128; off > 0; off >>= 1) {
        if (threadIdx.x < off) red[threadIdx.x] += red[threadIdx.x + off];
        __syncthreads();
    }
    if (threadIdx.x == 0) out[0] = -red[0] / (float)A_N;
}

// ---------------- launcher ----------------
extern "C" void kernel_launch(void* const* d_in, const int* in_sizes, int n_in,
                              void* d_out, int out_size, void* d_ws, size_t ws_size,
                              hipStream_t stream) {
    const float* anchors    = (const float*)d_in[0];
    const int*   labels     = (const int*)d_in[1];
    const float* proto_mem  = (const float*)d_in[2];
    const int*   proto_mask = (const int*)d_in[3];
    const float* local_mem  = (const float*)d_in[4];
    float* out = (float*)d_out;

    char* w = (char*)d_ws;
    auto carve = [&](size_t bytes) -> char* {
        char* p = w;
        w += (bytes + 255) & ~(size_t)255;
        return p;
    };
    _Float16* anchors_h = (_Float16*)carve((size_t)A_N * D_K * 2);
    _Float16* protos_h  = (_Float16*)carve((size_t)P_PAD * D_K * 2);
    float* m      = (float*)carve((size_t)A_N * 4);
    float* Nacc   = (float*)carve((size_t)A_N * 4);
    float* possum = (float*)carve((size_t)A_N * 4);
    int* pclass   = (int*)carve((size_t)P_PAD * 4);
    int* pvalid   = (int*)carve((size_t)P_PAD * 4);
    float* poscnt = (float*)carve((size_t)C_N * 4);
    int* counts   = (int*)carve((size_t)C_N * 4);
    int* lists    = (int*)carve((size_t)C_N * A_N * 4);

    k_zero<<<dim3(17), 256, 0, stream>>>(Nacc, counts);
    k_cvt_anchors<<<dim3(A_N), 256, 0, stream>>>(anchors, anchors_h);
    k_cvt_protos<<<dim3(P_PAD), 256, 0, stream>>>(proto_mem, proto_mask, protos_h, pclass, pvalid);
    k_bucket<<<dim3(16), 256, 0, stream>>>(labels, counts, lists);
    k_poscnt<<<dim3(1), 32, 0, stream>>>(proto_mask, poscnt);
    k_protomax<<<dim3(A_N / 16), 32, 0, stream>>>(anchors_h, protos_h, pvalid, m);
    k_local<<<dim3(C_N, A_N / 16, ZCH), 32, 0, stream>>>(local_mem, anchors_h, counts, lists, m, Nacc);
    k_protopos<<<dim3(A_N / 16), 32, 0, stream>>>(anchors_h, protos_h, pvalid, pclass, labels, m, Nacc, possum);
    k_final<<<dim3(1), 256, 0, stream>>>(possum, labels, poscnt, out);
}
